// Attention_4930622456396
// MI455X (gfx1250) — compile-verified, matrix operand order checked by reading, not run on hardware
//
#include <hip/hip_runtime.h>

// Problem constants (match reference)
#define BB  4
#define NQ  512
#define NK  1024
#define DD  64
#define SPLIT 4                       // flash split-K factor over keys
#define KT_PER_SPLIT ((NK / 32) / SPLIT)

typedef __attribute__((ext_vector_type(16))) _Float16 v16h;
typedef __attribute__((ext_vector_type(8)))  _Float16 v8h;
typedef __attribute__((ext_vector_type(8)))  float    v8f;
typedef __attribute__((ext_vector_type(4)))  float    v4f;

__device__ __forceinline__ v16h cat8(v8h lo, v8h hi) {
    return __builtin_shufflevector(lo, hi, 0,1,2,3,4,5,6,7,8,9,10,11,12,13,14,15);
}

__device__ __forceinline__ v8f wmma_f16(v16h a, v16h b, v8f c) {
    // D = A(16x32 f16) * B(32x16 f16) + C(16x16 f32)
    return __builtin_amdgcn_wmma_f32_16x16x32_f16(
        /*neg_a=*/false, a, /*neg_b=*/false, b,
        /*c_mod=*/(short)0, c, /*reuse_a=*/false, /*reuse_b=*/false);
}

// LDS byte offset of a __shared__ object: flat LDS addresses carry the LDS
// offset in bits [31:0] (ISA 10.2 aperture mapping), so truncation works.
__device__ __forceinline__ unsigned lds_off(const void* p) {
    return (unsigned)(unsigned long long)p;
}

// CDNA5 async copy global -> LDS (ASYNCcnt-tracked), via inline asm since the
// clang builtin arity differs across toolchains.
__device__ __forceinline__ void async_g2lds_b128(unsigned lds_addr, const float* gaddr) {
    asm volatile("global_load_async_to_lds_b128 %0, %1, off"
                 :: "v"(lds_addr), "v"(gaddr) : "memory");
}
__device__ __forceinline__ void wait_asynccnt0() {
    asm volatile("s_wait_asynccnt 0" ::: "memory");
}

// ---------------------------------------------------------------------------
// Kernel 1: fused 2-layer MLP (Linear->ReLU->Linear) for both key and query
// paths, done with WMMA. One workgroup = 4 waves = 64 rows.
// ---------------------------------------------------------------------------
__global__ void __launch_bounds__(128)
mlp_wmma_kernel(const float* __restrict__ x1,   // [B*NK, 64]
                const float* __restrict__ x2,   // [B*NQ, 64]
                const float* __restrict__ Wk1, const float* __restrict__ bk1,
                const float* __restrict__ Wk2, const float* __restrict__ bk2,
                const float* __restrict__ Wq1, const float* __restrict__ bq1,
                const float* __restrict__ Wq2, const float* __restrict__ bq2,
                float* __restrict__ kbuf,       // [B*NK, 64]
                float* __restrict__ qbuf)       // [B*NQ, 64]
{
    __shared__ __align__(16) _Float16 ldsW1T[DD * DD];   // [n][k] transposed
    __shared__ __align__(16) _Float16 ldsW2T[DD * DD];   // [n][k] transposed
    __shared__ __align__(16) _Float16 ldsH[4][16 * DD];  // per-wave hidden

    const int tid  = threadIdx.x;
    const int lane = tid & 31;
    const int wave = tid >> 5;
    const int nl   = lane & 15;           // column within 16-wide tile
    const int kb   = (lane < 16) ? 0 : 8; // K-offset for 16-bit operand layout
    const int m    = lane & 15;           // A-fragment row

    // Select job: blocks [0,64) -> key path over x1, [64,96) -> query path.
    const int wg = blockIdx.x;
    const float *X, *W1, *B1, *W2, *B2;
    float* OUT;
    long row0;
    if (wg < (BB * NK) / 64) {
        X = x1; W1 = Wk1; B1 = bk1; W2 = Wk2; B2 = bk2; OUT = kbuf;
        row0 = (long)wg * 64;
    } else {
        X = x2; W1 = Wq1; B1 = bq1; W2 = Wq2; B2 = bq2; OUT = qbuf;
        row0 = (long)(wg - (BB * NK) / 64) * 64;
    }

    // Stage both weight matrices transposed into LDS (f32 -> f16).
    for (int i = tid; i < DD * DD; i += blockDim.x) {
        const int k = i >> 6, n = i & 63;
        ldsW1T[n * DD + k] = (_Float16)W1[i];
        ldsW2T[n * DD + k] = (_Float16)W2[i];
    }

    // Per-lane biases for the 4 column tiles.
    float b1v[4], b2v[4];
#pragma unroll
    for (int c = 0; c < 4; ++c) { b1v[c] = B1[c * 16 + nl]; b2v[c] = B2[c * 16 + nl]; }

    __syncthreads();

    // Build the 2(Kchunk) x 4(Ntile) B-fragments for both layers.
    v16h w1f[2][4], w2f[2][4];
#pragma unroll
    for (int c = 0; c < 4; ++c) {
        const _Float16* p1 = &ldsW1T[(c * 16 + nl) * DD];
        const _Float16* p2 = &ldsW2T[(c * 16 + nl) * DD];
#pragma unroll
        for (int kc = 0; kc < 2; ++kc) {
            w1f[kc][c] = cat8(*(const v8h*)(p1 + kc * 32 + kb),
                              *(const v8h*)(p1 + kc * 32 + kb + 16));
            w2f[kc][c] = cat8(*(const v8h*)(p2 + kc * 32 + kb),
                              *(const v8h*)(p2 + kc * 32 + kb + 16));
        }
    }

    // A-fragments of x for this wave's 16 rows.
    const long rowm = row0 + (long)wave * 16 + m;
    const float* xp = X + rowm * DD;
    v16h ax[2];
#pragma unroll
    for (int kc = 0; kc < 2; ++kc) {
        v4f f0 = *(const v4f*)(xp + kc * 32 + kb);
        v4f f1 = *(const v4f*)(xp + kc * 32 + kb + 4);
        v4f f2 = *(const v4f*)(xp + kc * 32 + kb + 16);
        v4f f3 = *(const v4f*)(xp + kc * 32 + kb + 20);
        v16h a;
#pragma unroll
        for (int j = 0; j < 4; ++j) {
            a[j]      = (_Float16)f0[j];
            a[4 + j]  = (_Float16)f1[j];
            a[8 + j]  = (_Float16)f2[j];
            a[12 + j] = (_Float16)f3[j];
        }
        ax[kc] = a;
    }

    // Layer 1: h = relu(x @ W1 + b1); write back to LDS in row-major f16.
    _Float16* hb = &ldsH[wave][0];
#pragma unroll
    for (int c = 0; c < 4; ++c) {
        v8f cc;
#pragma unroll
        for (int g = 0; g < 8; ++g) cc[g] = b1v[c];
        cc = wmma_f16(ax[0], w1f[0][c], cc);
        cc = wmma_f16(ax[1], w1f[1][c], cc);
#pragma unroll
        for (int g = 0; g < 8; ++g) {
            const int rl = (lane < 16) ? g : (8 + g);     // C-layout row
            hb[rl * DD + c * 16 + nl] = (_Float16)fmaxf(cc[g], 0.0f);
        }
    }
    __syncthreads();

    // Layer 2 A-fragments from the staged hidden activations.
    const _Float16* hp = hb + m * DD;
    v16h ah[2];
#pragma unroll
    for (int kc = 0; kc < 2; ++kc)
        ah[kc] = cat8(*(const v8h*)(hp + kc * 32 + kb),
                      *(const v8h*)(hp + kc * 32 + kb + 16));

    // Layer 2: out = h @ W2 + b2, stored f32.
#pragma unroll
    for (int c = 0; c < 4; ++c) {
        v8f cc;
#pragma unroll
        for (int g = 0; g < 8; ++g) cc[g] = b2v[c];
        cc = wmma_f16(ah[0], w2f[0][c], cc);
        cc = wmma_f16(ah[1], w2f[1][c], cc);
#pragma unroll
        for (int g = 0; g < 8; ++g) {
            const int rl = (lane < 16) ? g : (8 + g);
            OUT[(row0 + (long)wave * 16 + rl) * DD + c * 16 + nl] = cc[g];
        }
    }
}

// ---------------------------------------------------------------------------
// Kernel 2: flash-style Laplace attention with split-K over keys.
//   S = -L1(k,q) (VALU) -> online softmax -> P @ V via WMMA.
// Grid = B * NQ/64 * SPLIT workgroups; 4 waves x 16 query rows per WG.
// Each split writes an unnormalized accumulator + its (m, l) state.
// ---------------------------------------------------------------------------
__global__ void __launch_bounds__(128)
laplace_attn_kernel(const float* __restrict__ kbuf,  // [B, NK, 64]
                    const float* __restrict__ qbuf,  // [B, NQ, 64]
                    const float* __restrict__ vbuf,  // [B, NK, 64] (= r)
                    float* __restrict__ pO,          // [SPLIT, B*NQ, 64]
                    float* __restrict__ pML)         // [SPLIT, B*NQ, 2]
{
    __shared__ __align__(16) float     ldsK[32 * DD];   // [key][d] f32
    __shared__ __align__(16) _Float16  ldsVT[DD * 32];  // [d][key] f16 (transposed)

    const int tid  = threadIdx.x;
    const int lane = tid & 31;
    const int wave = tid >> 5;
    const int nl   = lane & 15;
    const int kb   = (lane < 16) ? 0 : 8;
    const int m    = lane & 15;

    const int sp = blockIdx.x & (SPLIT - 1);
    const int bq = blockIdx.x >> 2;          // 0..31
    const int b  = bq >> 3;                  // NQ/64 = 8 q-tiles per batch
    const int qt = bq & 7;
    const int qrow0 = qt * 64 + wave * 16;

    const float* Kp = kbuf + (long)b * NK * DD;
    const float* Vp = vbuf + (long)b * NK * DD;
    const unsigned ldsK_base = lds_off(&ldsK[0]);

    // This lane's query row, kept in registers (f32 for exact L1 distance).
    const float* qp = qbuf + ((long)b * NQ + qrow0 + m) * DD;
    v4f qv[16];
#pragma unroll
    for (int i = 0; i < 16; ++i) qv[i] = *(const v4f*)(qp + i * 4);

    v8f acc[4];
#pragma unroll
    for (int c = 0; c < 4; ++c)
#pragma unroll
        for (int g = 0; g < 8; ++g) acc[c][g] = 0.0f;
    float run_m = -1.0e30f;
    float run_l = 0.0f;

    for (int kt = sp * KT_PER_SPLIT; kt < (sp + 1) * KT_PER_SPLIT; ++kt) {
        const float* ktile = Kp + (long)kt * 32 * DD;
        const float* vtile = Vp + (long)kt * 32 * DD;

        // K tile: CDNA5 async copy global -> LDS (no VGPR round trip).
        for (int i = tid; i < 512; i += 128)               // 512 x b128
            async_g2lds_b128(ldsK_base + (unsigned)i * 16, ktile + i * 4);

        // V tile: f32 -> f16, transposed [d][key], via VALU.
        for (int i = tid; i < 2048; i += 128) {
            const int key = i >> 6, d = i & 63;
            ldsVT[d * 32 + key] = (_Float16)vtile[i];
        }
        if (kt + 1 < (sp + 1) * KT_PER_SPLIT) {            // prefetch next tile
            __builtin_prefetch(ktile + 32 * DD + tid * 16, 0, 0);
            __builtin_prefetch(vtile + 32 * DD + tid * 16, 0, 0);
        }
        wait_asynccnt0();
        __syncthreads();

        // Each lane computes the 16 score values its A-fragment slot needs:
        // row = lane&15, key = kb + (h<8 ? h : h+8)  (16-bit A operand layout).
        float s[16];
#pragma unroll
        for (int h = 0; h < 16; ++h) {
            const int key = kb + ((h < 8) ? h : h + 8);
            const float* kp = &ldsK[key * DD];
            float a0 = 0.f, a1 = 0.f, a2 = 0.f, a3 = 0.f;
#pragma unroll
            for (int i = 0; i < 16; ++i) {
                v4f kv = *(const v4f*)(kp + i * 4);
                v4f dq = qv[i] - kv;
                a0 += __builtin_fabsf(dq.x);
                a1 += __builtin_fabsf(dq.y);
                a2 += __builtin_fabsf(dq.z);
                a3 += __builtin_fabsf(dq.w);
            }
            s[h] = -((a0 + a1) + (a2 + a3));
        }

        // Online softmax: per-row max/sum = in-lane reduce + swap halves.
        float mt = s[0];
#pragma unroll
        for (int h = 1; h < 16; ++h) mt = fmaxf(mt, s[h]);
        mt = fmaxf(mt, __shfl_xor(mt, 16, 32));
        const float newm  = fmaxf(run_m, mt);
        const float alpha = __expf(run_m - newm);
        run_m = newm;

        v16h pfrag;
        float ts = 0.f;
#pragma unroll
        for (int h = 0; h < 16; ++h) {
            const float p = __expf(s[h] - newm);
            ts += p;
            pfrag[h] = (_Float16)p;
        }
        ts += __shfl_xor(ts, 16, 32);
        run_l = run_l * alpha + ts;

        // Rescale accumulators: row r's alpha lives in lane r.
#pragma unroll
        for (int g = 0; g < 8; ++g) {
            const int src = ((lane < 16) ? g : (8 + g));
            const float ag = __shfl(alpha, src, 32);
#pragma unroll
            for (int c = 0; c < 4; ++c) acc[c][g] *= ag;
        }

        // P @ V with WMMA: B-fragments straight out of the transposed V tile.
#pragma unroll
        for (int c = 0; c < 4; ++c) {
            const _Float16* vp = &ldsVT[(c * 16 + nl) * 32];
            v16h bf = cat8(*(const v8h*)(vp + kb),
                           *(const v8h*)(vp + kb + 16));
            acc[c] = wmma_f16(pfrag, bf, acc[c]);
        }
        __syncthreads();   // LDS reused next tile
    }

    // Store unnormalized partial accumulator + per-row (m, l) state.
    const long prow0 = (long)sp * (BB * NQ) + (long)b * NQ + qrow0;
    float* po = pO + prow0 * DD;
#pragma unroll
    for (int c = 0; c < 4; ++c)
#pragma unroll
        for (int g = 0; g < 8; ++g) {
            const int rr = (lane < 16) ? g : (8 + g);
            po[(long)rr * DD + c * 16 + nl] = acc[c][g];
        }
    if (lane < 16) {
        pML[(prow0 + m) * 2 + 0] = run_m;
        pML[(prow0 + m) * 2 + 1] = run_l;
    }
}

// ---------------------------------------------------------------------------
// Kernel 3: merge the SPLIT partials:
//   out = sum_s e^{m_s-M} O_s / sum_s e^{m_s-M} l_s ,  M = max_s m_s
// ---------------------------------------------------------------------------
__global__ void __launch_bounds__(256)
splitk_combine_kernel(const float* __restrict__ pO,
                      const float* __restrict__ pML,
                      float* __restrict__ out)
{
    const int idx = blockIdx.x * 256 + threadIdx.x;   // over B*NQ*64
    const int row = idx >> 6;
    const int col = idx & 63;

    float mv[SPLIT], lv[SPLIT];
    float M = -1.0e30f;
#pragma unroll
    for (int s = 0; s < SPLIT; ++s) {
        mv[s] = pML[((long)s * (BB * NQ) + row) * 2 + 0];
        lv[s] = pML[((long)s * (BB * NQ) + row) * 2 + 1];
        M = fmaxf(M, mv[s]);
    }
    float denom = 0.f, o = 0.f;
#pragma unroll
    for (int s = 0; s < SPLIT; ++s) {
        const float w = __expf(mv[s] - M);
        denom += w * lv[s];
        o += w * pO[((long)s * (BB * NQ) + row) * DD + col];
    }
    out[idx] = o / denom;
}

// ---------------------------------------------------------------------------
extern "C" void kernel_launch(void* const* d_in, const int* in_sizes, int n_in,
                              void* d_out, int out_size, void* d_ws, size_t ws_size,
                              hipStream_t stream) {
    const float* x1  = (const float*)d_in[0];
    const float* x2  = (const float*)d_in[1];
    const float* r   = (const float*)d_in[2];
    const float* Wk1 = (const float*)d_in[3];
    const float* bk1 = (const float*)d_in[4];
    const float* Wk2 = (const float*)d_in[5];
    const float* bk2 = (const float*)d_in[6];
    const float* Wq1 = (const float*)d_in[7];
    const float* bq1 = (const float*)d_in[8];
    const float* Wq2 = (const float*)d_in[9];
    const float* bq2 = (const float*)d_in[10];

    float* kbuf = (float*)d_ws;                               // [B*NK*D]
    float* qbuf = kbuf + (size_t)BB * NK * DD;                // [B*NQ*D]
    float* pO   = qbuf + (size_t)BB * NQ * DD;                // [SPLIT*B*NQ*D]
    float* pML  = pO   + (size_t)SPLIT * BB * NQ * DD;        // [SPLIT*B*NQ*2]

    // MLPs: 64 row-tiles for the key path + 32 for the query path.
    mlp_wmma_kernel<<<(BB * NK + BB * NQ) / 64, 128, 0, stream>>>(
        x1, x2, Wk1, bk1, Wk2, bk2, Wq1, bq1, Wq2, bq2, kbuf, qbuf);

    // Attention: B * NQ/64 * SPLIT workgroups (fills the WGP array).
    laplace_attn_kernel<<<BB * (NQ / 64) * SPLIT, 128, 0, stream>>>(
        kbuf, qbuf, r, pO, pML);

    // Merge split-K partials.
    splitk_combine_kernel<<<(BB * NQ * DD) / 256, 256, 0, stream>>>(
        pO, pML, (float*)d_out);
}
